// QuantLinear_81389630259465
// MI455X (gfx1250) — compile-verified, matrix operand order checked by reading, not run on hardware
//
#include <hip/hip_runtime.h>

// CDNA5 / gfx1250: wave32, WMMA 16x16x32 bf16 -> f32 accumulators.
typedef __attribute__((ext_vector_type(16))) __bf16 v16bf;
typedef __attribute__((ext_vector_type(8)))  float  v8f;

#define BM 128      // M rows per block
#define BN 128      // N cols per block
#define KT 32       // K per LDS stage (one WMMA k-step)
#define LDSTR 40    // LDS row stride in bf16 elems (32 data + 8 pad -> 80B rows)
#define OSTR  136   // epilogue C-tile row stride in bf16 elems (128 + 8 pad)
#define K_DIM 4096
#define N_DIM 11008
#define M_DIM 8192  // B*S = 4*2048
#define NK (K_DIM / KT)

// Hardware byte-select uint8 -> float converts (VOP1). Inline asm keeps them
// opaque so the optimizer can't fold them into a software i8->bf16 path.
__device__ __forceinline__ float cvt_ub0(unsigned int q) {
  float r; asm("v_cvt_f32_ubyte0 %0, %1" : "=v"(r) : "v"(q)); return r;
}
__device__ __forceinline__ float cvt_ub1(unsigned int q) {
  float r; asm("v_cvt_f32_ubyte1 %0, %1" : "=v"(r) : "v"(q)); return r;
}
__device__ __forceinline__ float cvt_ub2(unsigned int q) {
  float r; asm("v_cvt_f32_ubyte2 %0, %1" : "=v"(r) : "v"(q)); return r;
}
__device__ __forceinline__ float cvt_ub3(unsigned int q) {
  float r; asm("v_cvt_f32_ubyte3 %0, %1" : "=v"(r) : "v"(q)); return r;
}

// Packed fp32 -> bf16 hardware convert: D[15:0]=bf16(S0), D[31:16]=bf16(S1).
__device__ __forceinline__ unsigned int cvt_pk_bf16(float lo, float hi) {
  unsigned int r;
  asm("v_cvt_pk_bf16_f32 %0, %1, %2" : "=v"(r) : "v"(lo), "v"(hi));
  return r;
}

// 8 fp32 -> 8 bf16 (one b128's worth) via 4 packed converts.
__device__ __forceinline__ uint4 pack8(const v8f& f) {
  uint4 u;
  u.x = cvt_pk_bf16(f[0], f[1]);
  u.y = cvt_pk_bf16(f[2], f[3]);
  u.z = cvt_pk_bf16(f[4], f[5]);
  u.w = cvt_pk_bf16(f[6], f[7]);
  return u;
}

extern "C" __global__ __launch_bounds__(256)
void qlinear_wmma_kernel(const float* __restrict__ x,
                         const signed char* __restrict__ qw,
                         const float* __restrict__ scale,
                         const float* __restrict__ zp,
                         const float* __restrict__ bias,
                         __bf16* __restrict__ out) {
  // [buf][A|B][row*LDSTR]; reused after the K loop as the C staging tile.
  __shared__ __align__(16) unsigned short sMem[2][2][BM * LDSTR];
  __shared__ float sRow[2][BM];

  const int t    = threadIdx.x;
  const int lane = t & 31;
  const int wave = t >> 5;
  const int wm   = wave & 1;   // 2 waves along M (64 rows each)
  const int wn   = wave >> 1;  // 4 waves along N (32 cols each)
  const int blockN = blockIdx.x;
  const int blockM = blockIdx.y;

  // Staging assignment: thread t owns row t>>1 of the tile, half (t&1) of KT.
  const int ldRow  = t >> 1;  // 0..127
  const int ldHalf = t & 1;   // 0..1

  const float*       aBase = x  + (size_t)(blockM * BM + ldRow) * K_DIM + ldHalf * 16;
  const signed char* bBase = qw + (size_t)(blockN * BN + ldRow) * K_DIM + ldHalf * 16;

  v8f c[4][2];
#pragma unroll
  for (int i = 0; i < 4; ++i)
#pragma unroll
    for (int j = 0; j < 2; ++j)
#pragma unroll
      for (int r = 0; r < 8; ++r) c[i][j][r] = 0.0f;

  float rs = 0.0f;  // partial activation row-sum (for zero-point correction)

  v8f   ar[2];  // 16 fp32 activations for this thread's slice
  uint4 br;     // 16 int8 weights

  auto loadStage = [&](int kt) {
    const v8f* ap = (const v8f*)(aBase + (size_t)kt * KT);
    ar[0] = ap[0];
    ar[1] = ap[1];
    br = *(const uint4*)(bBase + (size_t)kt * KT);
    // L2-directed prefetch well ahead of the software pipeline
    if (kt + 8 < NK) {
      __builtin_prefetch(aBase + (size_t)(kt + 8) * KT, 0, 3);
      __builtin_prefetch(bBase + (size_t)(kt + 8) * KT, 0, 3);
    }
  };

  auto storeStage = [&](int p) {
    // A: fp32 -> bf16 via packed hardware converts; accumulate row-sum.
    uint4* da = (uint4*)&sMem[p][0][ldRow * LDSTR + ldHalf * 16];
    da[0] = pack8(ar[0]);
    da[1] = pack8(ar[1]);
#pragma unroll
    for (int r = 0; r < 8; ++r) rs += ar[0][r] + ar[1][r];
    // B: store u = q + 128 = q ^ 0x80 (exact in bf16); the +128 is folded
    // into the zero-point correction in the epilogue.
    uint4* db = (uint4*)&sMem[p][1][ldRow * LDSTR + ldHalf * 16];
    union { uint4 u; unsigned int w[4]; } bw;
    bw.u = br;
    v8f f0, f1;
    {
      const unsigned int q0 = bw.w[0] ^ 0x80808080u;
      const unsigned int q1 = bw.w[1] ^ 0x80808080u;
      const unsigned int q2 = bw.w[2] ^ 0x80808080u;
      const unsigned int q3 = bw.w[3] ^ 0x80808080u;
      f0[0] = cvt_ub0(q0); f0[1] = cvt_ub1(q0);
      f0[2] = cvt_ub2(q0); f0[3] = cvt_ub3(q0);
      f0[4] = cvt_ub0(q1); f0[5] = cvt_ub1(q1);
      f0[6] = cvt_ub2(q1); f0[7] = cvt_ub3(q1);
      f1[0] = cvt_ub0(q2); f1[1] = cvt_ub1(q2);
      f1[2] = cvt_ub2(q2); f1[3] = cvt_ub3(q2);
      f1[4] = cvt_ub0(q3); f1[5] = cvt_ub1(q3);
      f1[6] = cvt_ub2(q3); f1[7] = cvt_ub3(q3);
    }
    db[0] = pack8(f0);
    db[1] = pack8(f1);
  };

  // Load a 16-bit WMMA operand fragment per the CDNA5 VGPR layout:
  // lane<16 holds K {0..7,16..23}, lane>=16 holds K {8..15,24..31}.
  auto ldFrag = [&](const unsigned short* rowBase) -> v16bf {
    const int half8 = (lane >> 4) * 8;
    union { uint4 u[2]; v16bf v; } f;
    f.u[0] = *(const uint4*)(rowBase + half8);
    f.u[1] = *(const uint4*)(rowBase + 16 + half8);
    return f.v;
  };

  auto compute = [&](int p) {
    const int lm = lane & 15;
    v16bf a[4], b[2];
#pragma unroll
    for (int mt = 0; mt < 4; ++mt)
      a[mt] = ldFrag(&sMem[p][0][(wm * 64 + mt * 16 + lm) * LDSTR]);
#pragma unroll
    for (int nt = 0; nt < 2; ++nt)
      b[nt] = ldFrag(&sMem[p][1][(wn * 32 + nt * 16 + lm) * LDSTR]);
#pragma unroll
    for (int mt = 0; mt < 4; ++mt)
#pragma unroll
      for (int nt = 0; nt < 2; ++nt)
        c[mt][nt] = __builtin_amdgcn_wmma_f32_16x16x32_bf16(
            false, a[mt], false, b[nt], (short)0, c[mt][nt], false, false);
  };

  // Software pipeline: global->regs (kt+2) | regs->LDS (kt+1) | WMMA (kt)
  loadStage(0);
  storeStage(0);
  loadStage(1);
  for (int kt = 0; kt < NK; ++kt) {
    __syncthreads();  // buf[kt&1] ready; buf[(kt+1)&1] fully consumed
    if (kt + 1 < NK) storeStage((kt + 1) & 1);
    if (kt + 2 < NK) loadStage(kt + 2);
    compute(kt & 1);
  }

  // Combine the two half-row sums; sync also retires all LDS stage reads so
  // sMem can be reused as the C staging tile.
  sRow[ldHalf][ldRow] = rs;
  __syncthreads();

  unsigned short* sOut = &sMem[0][0][0];  // 128 x OSTR bf16 tile (34KB < 40KB)
  const int lm = lane & 15;
  const int lh = lane >> 4;
#pragma unroll
  for (int mt = 0; mt < 4; ++mt) {
    float rsum[8];
#pragma unroll
    for (int r = 0; r < 8; ++r) {
      const int mLocal = wm * 64 + mt * 16 + lh * 8 + r;
      rsum[r] = sRow[0][mLocal] + sRow[1][mLocal];
    }
#pragma unroll
    for (int nt = 0; nt < 2; ++nt) {
      const int nLocal = wn * 32 + nt * 16 + lm;
      const int nG = blockN * BN + nLocal;
      const float sc   = scale[nG];
      const float corr = sc * (zp[nG] + 128.0f);  // accounts for B' = q + 128
      const float bs   = bias[nG];
      v8f vals;
#pragma unroll
      for (int r = 0; r < 8; ++r)
        vals[r] = sc * c[mt][nt][r] - corr * rsum[r] + bs;
      union { uint4 u; unsigned short s[8]; } bv;
      bv.u = pack8(vals);
#pragma unroll
      for (int r = 0; r < 8; ++r) {
        const int mLocal = wm * 64 + mt * 16 + lh * 8 + r;
        sOut[mLocal * OSTR + nLocal] = bv.s[r];
      }
    }
  }
  __syncthreads();

  // Coalesced copy-out: each thread streams 64 contiguous bf16 (128B) per row
  // half via b128 stores.
  {
    const int r = t >> 1;   // 0..127
    const int h = t & 1;    // 0..1
    const uint4* src = (const uint4*)&sOut[r * OSTR + h * 64];
    uint4* dst = (uint4*)(out + (size_t)(blockM * BM + r) * N_DIM
                              + blockN * BN + h * 64);
#pragma unroll
    for (int i = 0; i < 8; ++i) dst[i] = src[i];
  }
}

extern "C" void kernel_launch(void* const* d_in, const int* in_sizes, int n_in,
                              void* d_out, int out_size, void* d_ws, size_t ws_size,
                              hipStream_t stream) {
  const float*       x  = (const float*)d_in[0];
  const signed char* qw = (const signed char*)d_in[1];
  const float*       sc = (const float*)d_in[2];
  const float*       zp = (const float*)d_in[3];
  const float*       bs = (const float*)d_in[4];
  __bf16*            out = (__bf16*)d_out;

  dim3 grid(N_DIM / BN, M_DIM / BM);  // 86 x 64 blocks
  qlinear_wmma_kernel<<<grid, dim3(256), 0, stream>>>(x, qw, sc, zp, bs, out);
}
